// NNBlendFM_68075231642206
// MI455X (gfx1250) — matrix operand order repaired and checked by reading, not compiled, not run on hardware
//
#include <hip/hip_runtime.h>
#include <hip/hip_bf16.h>
#include <math.h>

// Problem dims (match reference)
#define BATCH   8192
#define IN_SIZE 512
#define HIDDEN  2048
#define HEADS   64
#define RANK    16
#define NCOMB   1152   // 64 (lin) + 1024 (V rows), padded to multiple of 128
#define NDIAG   128    // 64 diag rows padded to 128

typedef __bf16 bf16_t;
typedef bf16_t v16bf __attribute__((ext_vector_type(16)));
typedef float  v8f   __attribute__((ext_vector_type(8)));
typedef unsigned int u32x4 __attribute__((ext_vector_type(4)));
typedef int i32x4 __attribute__((ext_vector_type(4)));
typedef int i32x8 __attribute__((ext_vector_type(8)));

#if __has_builtin(__builtin_amdgcn_tensor_load_to_lds) && __has_builtin(__builtin_amdgcn_s_wait_tensorcnt)
#define HAS_TDM 1
#else
#define HAS_TDM 0
#endif

// LDS tile: 128 rows x 32 bf16 (64B) padded to 80B row pitch (TDM pad: 4 DW per 16 DW)
#define TILE_PITCH 80

// ---------------- helpers ----------------

__device__ __forceinline__ bf16_t f32_to_bf16(float f) {
  union { float f; unsigned int u; } in; in.f = f;
  unsigned int u = in.u;
  u += 0x7fffu + ((u >> 16) & 1u);        // round-to-nearest-even
  union { unsigned short s; bf16_t b; } out;
  out.s = (unsigned short)(u >> 16);
  return out.b;
}

#if HAS_TDM
// TDM: DMA a 128x32 bf16 tile (row stride ldk elements) from global into LDS,
// writing 80B-pitch rows (pad_amount=4 DW every pad_interval=16 DW).
// D# layout per CDNA5 ISA ch.8 (group0: flags/lds/global addr, group1: dims).
__device__ __forceinline__ void tdm_load_128x32(const bf16_t* gsrc, unsigned lds_off, int ldk) {
  unsigned long long ga = (unsigned long long)(size_t)gsrc;
  unsigned kdim = (unsigned)ldk;
  u32x4 g0 = { 1u,                                        // count=1, no gather
               lds_off,                                   // lds_addr
               (unsigned)(ga & 0xffffffffull),            // global_addr[31:0]
               (unsigned)((ga >> 32) & 0x01ffffffull)     // global_addr[56:32]
                 | 0x80000000u };                         // type=2 ("image")
  i32x8 g1;
  g1[0] = (int)((1u << 16)     // data_size = 1 (2 bytes)
              | (1u << 20)     // pad_enable
              | (3u << 22)     // pad_interval code 3 -> every 16 DWORDs (one 64B row)
              | (3u << 25));   // pad_amount   code 3 -> 4 DWORDs (16B)
  g1[1] = (int)((kdim & 0xffffu) << 16);            // tensor_dim0[15:0]
  g1[2] = (int)((kdim >> 16) | (128u << 16));       // tensor_dim0[31:16] | tensor_dim1[15:0]=128
  g1[3] = (int)(32u << 16);                         // tensor_dim1 hi=0 | tile_dim0=32
  g1[4] = (int)128u;                                // tile_dim1=128, tile_dim2=0
  g1[5] = (int)kdim;                                // tensor_dim0_stride[31:0]
  g1[6] = 0;                                        // stride0 hi | stride1 lo (unused, 2D)
  g1[7] = 0;
  i32x4 z4 = { 0, 0, 0, 0 };
  i32x8 z8 = { 0, 0, 0, 0, 0, 0, 0, 0 };
  // 6-arg toolchain variant: (g0, g1, g2, g3, g4, cpol)
  __builtin_amdgcn_tensor_load_to_lds(g0, g1, z4, z4, z8, 0);
}
#endif

// Stage a 128x32 bf16 tile into LDS. TDM path: one designated wave issues the DMA
// (EXEC-independent, wave-level). Fallback: cooperative copy by all 256 threads.
template<int ISSUER>
__device__ __forceinline__ void stage_tile(const bf16_t* __restrict__ gsrc,
                                           char* lbuf, int ldk, int tid) {
#if HAS_TDM
  if ((tid >> 5) == ISSUER)
    tdm_load_128x32(gsrc, (unsigned)(size_t)lbuf, ldk);
#else
  for (int c = tid; c < 512; c += 256) {      // 128 rows x 4 chunks of 16B
    int r = c >> 2, q = c & 3;
    *reinterpret_cast<u32x4*>(lbuf + r * TILE_PITCH + q * 16) =
        *reinterpret_cast<const u32x4*>(gsrc + (size_t)r * ldk + q * 8);
  }
#endif
}

// A-fragment (16x32 bf16) from LDS tile (80B pitch).
// ISA layout: lane<16 -> M=lane, K in {0..7, 16..23}; lane>=16 -> M=lane-16, K in {8..15, 24..31}
__device__ __forceinline__ v16bf lds_fragA(const char* lbuf, int row0, int lane) {
  const char* p = lbuf + (row0 + (lane & 15)) * TILE_PITCH + (lane >> 4) * 16;
  union { v16bf v; u32x4 q[2]; } u;
  u.q[0] = *reinterpret_cast<const u32x4*>(p);        // K = c*8 .. +7
  u.q[1] = *reinterpret_cast<const u32x4*>(p + 32);   // K = 16+c*8 .. +7
  return u.v;
}

// B-fragment (32x16 bf16, K x N) from LDS tile holding Bt rows (one output col per row).
// ISA layout: lane<16 -> N=lane, K in {0..7, 16..23}; lane>=16 -> N=lane-16, K in {8..15, 24..31}
__device__ __forceinline__ v16bf lds_fragB(const char* lbuf, int col0, int lane) {
  const char* p = lbuf + (col0 + (lane & 15)) * TILE_PITCH + (lane >> 4) * 16;
  union { v16bf v; u32x4 q[2]; } u;
  u.q[0] = *reinterpret_cast<const u32x4*>(p);
  u.q[1] = *reinterpret_cast<const u32x4*>(p + 32);
  return u.v;
}

// ---------------- conversion / packing kernels ----------------

__global__ void convert_bf16_kernel(const float* __restrict__ in,
                                    bf16_t* __restrict__ out, int n) {
  int i = blockIdx.x * blockDim.x + threadIdx.x;
  if (i < n) out[i] = f32_to_bf16(in[i]);
}

// in: [R x C] f32 row-major  ->  out: [C x R] bf16 row-major (transposed)
__global__ void transpose_bf16_kernel(const float* __restrict__ in,
                                      bf16_t* __restrict__ out, int R, int C) {
  __shared__ float tile[32][33];
  int c0 = blockIdx.x * 32;
  int r0 = blockIdx.y * 32;
  int tx = threadIdx.x;           // 0..31
  for (int i = threadIdx.y; i < 32; i += 8)
    tile[i][tx] = in[(size_t)(r0 + i) * C + (c0 + tx)];
  __syncthreads();
  for (int i = threadIdx.y; i < 32; i += 8)
    out[(size_t)(c0 + i) * R + (r0 + tx)] = f32_to_bf16(tile[tx][i]);
}

// Build combined FM matrix Mc [NCOMB x HIDDEN] (rows: fm_w, then fm_V flat, then 0)
// and Dm [NDIAG x HIDDEN] (row h<64: sum_r V[h,r,d]^2, else 0), both bf16.
__global__ void build_fm_kernel(const float* __restrict__ fm_w,
                                const float* __restrict__ fm_V,
                                bf16_t* __restrict__ Mc,
                                bf16_t* __restrict__ Dm) {
  int d   = blockIdx.x * blockDim.x + threadIdx.x;   // 0..HIDDEN-1
  int row = blockIdx.y;                              // 0..NCOMB-1
  if (d >= HIDDEN) return;
  float v = 0.0f;
  if (row < HEADS)            v = fm_w[(size_t)row * HIDDEN + d];
  else if (row < HEADS + HEADS * RANK)
                              v = fm_V[(size_t)(row - HEADS) * HIDDEN + d];
  Mc[(size_t)row * HIDDEN + d] = f32_to_bf16(v);
  if (row < NDIAG) {
    float s = 0.0f;
    if (row < HEADS) {
      for (int r = 0; r < RANK; ++r) {
        float t = fm_V[(size_t)(row * RANK + r) * HIDDEN + d];
        s += t * t;
      }
    }
    Dm[(size_t)row * HIDDEN + d] = f32_to_bf16(s);
  }
}

// ---------------- WMMA GEMM with TDM-staged, double-buffered LDS tiles ----------------
// C[M,N] = A[M,K] @ Bt[N,K]^T ; 256 threads = 8 waves, block tile 128x128,
// wave tile 64x32 (4x2 array of 16x16x32 bf16 WMMAs).
// Wave 0 DMAs the A tile, wave 1 the B tile; TENSORcnt + barrier publishes buffers.
// TANH=true : out = tanh(acc + bias[n]) -> bf16 (and optionally its square -> bf16)
// TANH=false: out = acc -> f32
template<bool TANH>
__global__ void __launch_bounds__(256) gemm_wmma_kernel(
    const bf16_t* __restrict__ A,
    const bf16_t* __restrict__ Bt,
    const float*  __restrict__ bias,
    bf16_t* __restrict__ outBf,
    bf16_t* __restrict__ outSq,
    float*  __restrict__ outF,
    int M, int N, int K) {
  (void)M;
  __shared__ alignas(16) char lA[2][128 * TILE_PITCH];
  __shared__ alignas(16) char lB[2][128 * TILE_PITCH];

  const int tid  = threadIdx.x;
  const int lane = tid & 31;
  const int wave = tid >> 5;      // 0..7
  const int wm = wave & 1;        // M direction (2)
  const int wn = wave >> 1;       // N direction (4)
  const int mBlk = blockIdx.x * 128;
  const int nBlk = blockIdx.y * 128;
  const int nk = K >> 5;

  stage_tile<0>(A  + (size_t)mBlk * K, lA[0], K, tid);
  stage_tile<1>(Bt + (size_t)nBlk * K, lB[0], K, tid);

  v8f acc[4][2] = {};

  for (int kt = 0; kt < nk; ++kt) {
    const int cur = kt & 1;
    if (kt + 1 < nk) {
      stage_tile<0>(A  + (size_t)mBlk * K + (kt + 1) * 32, lA[cur ^ 1], K, tid);
      stage_tile<1>(Bt + (size_t)nBlk * K + (kt + 1) * 32, lB[cur ^ 1], K, tid);
#if HAS_TDM
      if (wave < 2) __builtin_amdgcn_s_wait_tensorcnt((short)1);  // current tile done
#endif
    } else {
#if HAS_TDM
      if (wave < 2) __builtin_amdgcn_s_wait_tensorcnt((short)0);
#endif
    }
    __syncthreads();

    v16bf a[4], b[2];
#pragma unroll
    for (int i = 0; i < 4; ++i) a[i] = lds_fragA(lA[cur], wm * 64 + 16 * i, lane);
#pragma unroll
    for (int j = 0; j < 2; ++j) b[j] = lds_fragB(lB[cur], wn * 32 + 16 * j, lane);
#pragma unroll
    for (int i = 0; i < 4; ++i)
#pragma unroll
      for (int j = 0; j < 2; ++j)
        acc[i][j] = __builtin_amdgcn_wmma_f32_16x16x32_bf16(
            false, a[i], false, b[j], (short)0, acc[i][j], false, false);

    __syncthreads();   // safe to overwrite buffer `cur` next-next iteration
  }

  // Epilogue. C/D layout: VGPR t, lanes 0-15 -> M=t, lanes 16-31 -> M=t+8; N=lane&15.
  const int ln = lane & 15;
  const int lh = (lane >> 4) << 3;        // 0 or 8
#pragma unroll
  for (int i = 0; i < 4; ++i) {
#pragma unroll
    for (int j = 0; j < 2; ++j) {
      const int n = nBlk + wn * 32 + j * 16 + ln;
      float bv = 0.0f;
      if (TANH) bv = bias[n];
#pragma unroll
      for (int t = 0; t < 8; ++t) {
        const int m = mBlk + wm * 64 + i * 16 + lh + t;
        float v = acc[i][j][t];
        if (TANH) {
          v = tanhf(v + bv);
          outBf[(size_t)m * N + n] = f32_to_bf16(v);
          if (outSq) outSq[(size_t)m * N + n] = f32_to_bf16(v * v);
        } else {
          outF[(size_t)m * N + n] = v;
        }
      }
    }
  }
}

// ---------------- FM epilogue ----------------
// out[h,b] = w0[h] + G1[b,h] + 0.5*(sum_r G1[b,64+16h+r]^2 - G2[b,h])
__global__ void fm_epilogue_kernel(const float* __restrict__ G1,
                                   const float* __restrict__ G2,
                                   const float* __restrict__ fm_w0,
                                   float* __restrict__ out) {
  int b = blockIdx.x * blockDim.x + threadIdx.x;
  int h = blockIdx.y;
  if (b >= BATCH) return;
  const float* g = G1 + (size_t)b * NCOMB;
  float lin  = g[h];
  float diag = G2[(size_t)b * NDIAG + h];
  float q = 0.0f;
#pragma unroll
  for (int r = 0; r < RANK; ++r) {
    float v = g[HEADS + h * RANK + r];
    q += v * v;
  }
  out[(size_t)h * BATCH + b] = fm_w0[h] + lin + 0.5f * (q - diag);
}

// ---------------- workspace layout (bytes) ----------------
static constexpr size_t W1T_OFF = 0;
static constexpr size_t W1T_SZ  = (size_t)HIDDEN * IN_SIZE * 2;
static constexpr size_t W2T_OFF = W1T_OFF + W1T_SZ;
static constexpr size_t W2T_SZ  = (size_t)HIDDEN * HIDDEN * 2;
static constexpr size_t W3T_OFF = W2T_OFF + W2T_SZ;
static constexpr size_t W3T_SZ  = (size_t)HIDDEN * HIDDEN * 2;
static constexpr size_t XB_OFF  = W3T_OFF + W3T_SZ;
static constexpr size_t XB_SZ   = (size_t)BATCH * IN_SIZE * 2;
static constexpr size_t MC_OFF  = XB_OFF + XB_SZ;
static constexpr size_t MC_SZ   = (size_t)NCOMB * HIDDEN * 2;
static constexpr size_t DM_OFF  = MC_OFF + MC_SZ;
static constexpr size_t DM_SZ   = (size_t)NDIAG * HIDDEN * 2;
static constexpr size_t H1_OFF  = DM_OFF + DM_SZ;
static constexpr size_t H_SZ    = (size_t)BATCH * HIDDEN * 2;
static constexpr size_t H2_OFF  = H1_OFF + H_SZ;
static constexpr size_t H3_OFF  = H2_OFF + H_SZ;
static constexpr size_t HS_OFF  = H3_OFF + H_SZ;
static constexpr size_t G1_OFF  = HS_OFF + H_SZ;
static constexpr size_t G1_SZ   = (size_t)BATCH * NCOMB * 4;
static constexpr size_t G2_OFF  = G1_OFF + G1_SZ;

extern "C" void kernel_launch(void* const* d_in, const int* in_sizes, int n_in,
                              void* d_out, int out_size, void* d_ws, size_t ws_size,
                              hipStream_t stream) {
  (void)in_sizes; (void)n_in; (void)out_size; (void)ws_size;
  const float* x   = (const float*)d_in[0];
  const float* W1  = (const float*)d_in[1];
  const float* b1  = (const float*)d_in[2];
  const float* W2  = (const float*)d_in[3];
  const float* b2  = (const float*)d_in[4];
  const float* W3  = (const float*)d_in[5];
  const float* b3  = (const float*)d_in[6];
  const float* fw0 = (const float*)d_in[7];
  const float* fw  = (const float*)d_in[8];
  const float* fV  = (const float*)d_in[9];
  float* out = (float*)d_out;
  char* ws = (char*)d_ws;

  bf16_t* W1t = (bf16_t*)(ws + W1T_OFF);
  bf16_t* W2t = (bf16_t*)(ws + W2T_OFF);
  bf16_t* W3t = (bf16_t*)(ws + W3T_OFF);
  bf16_t* xb  = (bf16_t*)(ws + XB_OFF);
  bf16_t* Mc  = (bf16_t*)(ws + MC_OFF);
  bf16_t* Dm  = (bf16_t*)(ws + DM_OFF);
  bf16_t* H1  = (bf16_t*)(ws + H1_OFF);
  bf16_t* H2  = (bf16_t*)(ws + H2_OFF);
  bf16_t* H3  = (bf16_t*)(ws + H3_OFF);
  bf16_t* Hs  = (bf16_t*)(ws + HS_OFF);
  float*  G1  = (float*)(ws + G1_OFF);
  float*  G2  = (float*)(ws + G2_OFF);

  // 1) convert x to bf16
  {
    int n = BATCH * IN_SIZE;
    convert_bf16_kernel<<<(n + 255) / 256, 256, 0, stream>>>(x, xb, n);
  }
  // 2) transpose+convert weights to [N x K] bf16
  {
    dim3 blk(32, 8);
    transpose_bf16_kernel<<<dim3(HIDDEN / 32, IN_SIZE / 32), blk, 0, stream>>>(W1, W1t, IN_SIZE, HIDDEN);
    transpose_bf16_kernel<<<dim3(HIDDEN / 32, HIDDEN / 32), blk, 0, stream>>>(W2, W2t, HIDDEN, HIDDEN);
    transpose_bf16_kernel<<<dim3(HIDDEN / 32, HIDDEN / 32), blk, 0, stream>>>(W3, W3t, HIDDEN, HIDDEN);
  }
  // 3) build FM matrices
  build_fm_kernel<<<dim3(HIDDEN / 256, NCOMB), 256, 0, stream>>>(fw, fV, Mc, Dm);

  // 4) three tanh GEMM layers (layer 3 also emits h^2 in bf16)
  gemm_wmma_kernel<true><<<dim3(BATCH / 128, HIDDEN / 128), 256, 0, stream>>>(
      xb, W1t, b1, H1, nullptr, nullptr, BATCH, HIDDEN, IN_SIZE);
  gemm_wmma_kernel<true><<<dim3(BATCH / 128, HIDDEN / 128), 256, 0, stream>>>(
      H1, W2t, b2, H2, nullptr, nullptr, BATCH, HIDDEN, HIDDEN);
  gemm_wmma_kernel<true><<<dim3(BATCH / 128, HIDDEN / 128), 256, 0, stream>>>(
      H2, W3t, b3, H3, Hs, nullptr, BATCH, HIDDEN, HIDDEN);

  // 5) FM GEMMs: G1 = H3 @ Mc^T, G2 = H3^2 @ Dm^T
  gemm_wmma_kernel<false><<<dim3(BATCH / 128, NCOMB / 128), 256, 0, stream>>>(
      H3, Mc, nullptr, nullptr, nullptr, G1, BATCH, NCOMB, HIDDEN);
  gemm_wmma_kernel<false><<<dim3(BATCH / 128, NDIAG / 128), 256, 0, stream>>>(
      Hs, Dm, nullptr, nullptr, nullptr, G2, BATCH, NDIAG, HIDDEN);

  // 6) combine
  fm_epilogue_kernel<<<dim3(BATCH / 256, HEADS), 256, 0, stream>>>(G1, G2, fw0, out);
}
